// SetAbstraction_22531398435389
// MI455X (gfx1250) — matrix operand order repaired and checked by reading, not compiled, not run on hardware
//
#include <hip/hip_runtime.h>

#define B_       8
#define N_       4096
#define C_       64
#define H_       128
#define NPOINT_  1024
#define NSAMPLE_ 32
#define R2_      0.0225f

typedef __attribute__((ext_vector_type(16))) __bf16    v16bf;
typedef __attribute__((ext_vector_type(8)))  float     v8f;
typedef __attribute__((ext_vector_type(4)))  unsigned  u32x4;

union Frag16  { u32x4 q[2]; v16bf v; };
union FragAcc { v8f v; float f[8]; };

__device__ __forceinline__ unsigned short f2bf(float x) {
  union { float f; unsigned u; } v; v.f = x;
  unsigned r = v.u + 0x7fffu + ((v.u >> 16) & 1u);
  return (unsigned short)(r >> 16);
}
__device__ __forceinline__ float bf2f(unsigned short h) {
  union { unsigned u; float f; } v; v.u = ((unsigned)h) << 16;
  return v.f;
}
__device__ __forceinline__ unsigned fbits(float x) {
  union { float f; unsigned u; } v; v.f = x; return v.u;
}

// ---------------- weight prep: f32 [K][128] -> bf16 [128][Kpad] (K-contiguous) -------------
__global__ void wprep_kernel(const float* __restrict__ W, unsigned short* __restrict__ WT,
                             int Kreal, int Kpad) {
  int i = blockIdx.x * blockDim.x + threadIdx.x;
  if (i >= H_ * Kpad) return;
  int n = i / Kpad;
  int k = i - n * Kpad;
  float v = (k < Kreal) ? W[k * H_ + n] : 0.0f;
  WT[i] = f2bf(v);
}

// ---------------- FPS: one block per batch, dist in regs, xyz in LDS -----------------------
__global__ void __launch_bounds__(256) fps_kernel(const float* __restrict__ xyz,
                                                  float* __restrict__ newxyz) {
  __shared__ __align__(16) float sx[N_ * 3];
  __shared__ unsigned long long wred[8];
  __shared__ int sfar;
  const int tid = threadIdx.x;
  const int b = blockIdx.x;
  const float* X = xyz + (size_t)b * N_ * 3;
  for (int i = tid; i < N_ * 3; i += 256) sx[i] = X[i];
  __syncthreads();
  const float c0x = sx[0], c0y = sx[1], c0z = sx[2];
  float dist[16];
#pragma unroll
  for (int j = 0; j < 16; ++j) {
    int p = tid + 256 * j;
    float dx = sx[p*3+0]-c0x, dy = sx[p*3+1]-c0y, dz = sx[p*3+2]-c0z;
    dist[j] = dx*dx + dy*dy + dz*dz;
  }
  if (tid == 0) {
    float* o = newxyz + (size_t)b * NPOINT_ * 3;
    o[0] = c0x; o[1] = c0y; o[2] = c0z;
  }
  for (int it = 1; it < NPOINT_; ++it) {
    unsigned long long best = 0ull;
#pragma unroll
    for (int j = 0; j < 16; ++j) {
      int p = tid + 256 * j;
      // max dist wins; ties -> smaller index (matches jnp.argmax first-max)
      unsigned long long key = (((unsigned long long)fbits(dist[j])) << 32) |
                               (unsigned long long)(unsigned)(N_ - 1 - p);
      best = key > best ? key : best;
    }
#pragma unroll
    for (int off = 16; off >= 1; off >>= 1) {
      unsigned long long o = __shfl_xor(best, off, 32);
      best = o > best ? o : best;
    }
    if ((tid & 31) == 0) wred[tid >> 5] = best;
    __syncthreads();
    if (tid == 0) {
      unsigned long long m = wred[0];
      for (int w = 1; w < 8; ++w) m = wred[w] > m ? wred[w] : m;
      sfar = N_ - 1 - (int)(unsigned)(m & 0xffffffffull);
    }
    __syncthreads();
    const int far = sfar;
    const float fx = sx[far*3+0], fy = sx[far*3+1], fz = sx[far*3+2];
    if (tid == 0) {
      float* o = newxyz + ((size_t)b * NPOINT_ + it) * 3;
      o[0] = fx; o[1] = fy; o[2] = fz;
    }
#pragma unroll
    for (int j = 0; j < 16; ++j) {
      int p = tid + 256 * j;
      float dx = sx[p*3+0]-fx, dy = sx[p*3+1]-fy, dz = sx[p*3+2]-fz;
      float d = dx*dx + dy*dy + dz*dz;
      dist[j] = d < dist[j] ? d : dist[j];
    }
  }
}

// ------------- ball query: stable argsort-take-32 via (bits(d2),idx) packed keys -----------
__global__ void __launch_bounds__(256) ballq_kernel(const float* __restrict__ xyz,
                                                    const float* __restrict__ newxyz,
                                                    unsigned* __restrict__ gidx) {
  __shared__ unsigned long long wred[8];
  __shared__ unsigned long long ssel;
  const int tid = threadIdx.x;
  const int grp = blockIdx.x;            // b*NPOINT + p
  const int b = grp >> 10;
  const float* X = xyz + (size_t)b * N_ * 3;
  const float cx = newxyz[(size_t)grp*3+0];
  const float cy = newxyz[(size_t)grp*3+1];
  const float cz = newxyz[(size_t)grp*3+2];
  unsigned long long key[16];
#pragma unroll
  for (int j = 0; j < 16; ++j) {
    int p = tid + 256 * j;
    float dx = X[p*3+0]-cx, dy = X[p*3+1]-cy, dz = X[p*3+2]-cz;
    float d2 = dx*dx + dy*dy + dz*dz;
    unsigned fb = (d2 <= R2_) ? fbits(d2) : 0x7f800000u;   // +inf if outside radius
    key[j] = (((unsigned long long)fb) << 32) | (unsigned long long)(unsigned)p;
  }
  unsigned long long lo = 0ull;
  for (int s = 0; s < NSAMPLE_; ++s) {
    unsigned long long best = ~0ull;
#pragma unroll
    for (int j = 0; j < 16; ++j) {
      unsigned long long k = key[j];
      if (k >= lo && k < best) best = k;
    }
#pragma unroll
    for (int off = 16; off >= 1; off >>= 1) {
      unsigned long long o = __shfl_xor(best, off, 32);
      best = o < best ? o : best;
    }
    if ((tid & 31) == 0) wred[tid >> 5] = best;
    __syncthreads();
    if (tid == 0) {
      unsigned long long m = wred[0];
      for (int w = 1; w < 8; ++w) m = wred[w] < m ? wred[w] : m;
      ssel = m;
      gidx[(size_t)grp * NSAMPLE_ + s] = (unsigned)(m & 0xffffffffull);
    }
    __syncthreads();
    lo = ssel + 1ull;
  }
}

// ---------------- WMMA fragment helpers (CDNA5 16x16x32 bf16 layouts) ----------------------
// A 16x32 bf16: lane l holds row M=(l&15); lanes<16: K octets {0..7,16..23}; lanes>=16: {8..15,24..31}
__device__ __forceinline__ v16bf load_a(const unsigned short* Xs, int ks, int m0, int k0, int lane) {
  const int m  = m0 + (lane & 15);
  const int kh = (lane >> 4) << 3;                 // 0 or 8
  const unsigned short* p = Xs + m * ks + k0 + kh;
  Frag16 f;
  f.q[0] = *(const u32x4*)(p);                      // K = k0+kh .. +7
  f.q[1] = *(const u32x4*)(p + 16);                 // K = k0+kh+16 .. +23
  return f.v;
}
// B 32x16 bf16 from WT [128][Kp] (K contiguous): lane holds col N=(l&15), K half by lane>=16
__device__ __forceinline__ v16bf load_b(const unsigned short* Wt, int Kp, int n0, int k0, int lane) {
  const int n  = n0 + (lane & 15);
  const int kh = (lane >> 4) << 3;                 // 0 or 8
  const unsigned short* p = Wt + n * Kp + k0 + kh;
  Frag16 f;
  f.q[0] = *(const u32x4*)(p);
  f.q[1] = *(const u32x4*)(p + 16);
  return f.v;
}
__device__ __forceinline__ v8f wmma_bf16(v16bf a, v16bf b, v8f c) {
  return __builtin_amdgcn_wmma_f32_16x16x32_bf16(false, a, false, b, (short)0, c, false, false);
}

// GEMM 2x2-tile: [32 x Kp](LDS bf16) @ [Kp x 128](global bf16, N-major) + bias -> relu -> LDS
// Four independent accumulator chains per iteration hide WMMA hazards; A frags feed 2 WMMAs.
__device__ __forceinline__ void gemm_relu(const unsigned short* Xs, const unsigned short* Wt,
                                          const float* __restrict__ bias, unsigned short* Out,
                                          int Kp, int KT, int lane) {
  const int nlo = lane & 15;
  const int mb  = (lane >> 4) << 3;                // 0 or 8 (C/D layout)
  for (int nt = 0; nt < 8; nt += 2) {
    const int n0 = nt * 16 + nlo;
    const int n1 = n0 + 16;
    const float bv0 = bias[n0];
    const float bv1 = bias[n1];
    FragAcc a00, a10, a01, a11;
#pragma unroll
    for (int r = 0; r < 8; ++r) { a00.f[r]=0.f; a10.f[r]=0.f; a01.f[r]=0.f; a11.f[r]=0.f; }
#pragma unroll
    for (int kt = 0; kt < KT; ++kt) {
      v16bf x0 = load_a(Xs, Kp, 0,  kt * 32, lane);
      v16bf x1 = load_a(Xs, Kp, 16, kt * 32, lane);
      v16bf b0 = load_b(Wt, Kp, nt * 16,      kt * 32, lane);
      v16bf b1 = load_b(Wt, Kp, nt * 16 + 16, kt * 32, lane);
      a00.v = wmma_bf16(x0, b0, a00.v);
      a10.v = wmma_bf16(x1, b0, a10.v);
      a01.v = wmma_bf16(x0, b1, a01.v);
      a11.v = wmma_bf16(x1, b1, a11.v);
    }
#pragma unroll
    for (int r = 0; r < 8; ++r) {
      Out[(mb + r) * H_ + n0]      = f2bf(fmaxf(a00.f[r] + bv0, 0.f));
      Out[(16 + mb + r) * H_ + n0] = f2bf(fmaxf(a10.f[r] + bv0, 0.f));
      Out[(mb + r) * H_ + n1]      = f2bf(fmaxf(a01.f[r] + bv1, 0.f));
      Out[(16 + mb + r) * H_ + n1] = f2bf(fmaxf(a11.f[r] + bv1, 0.f));
    }
  }
}

// --------------- per-group MLP chain: one wave per (batch, centroid) -----------------------
__global__ void __launch_bounds__(32) sa_compute_kernel(
    const float* __restrict__ xyz, const float* __restrict__ feats,
    const float* __restrict__ newxyz, const unsigned* __restrict__ gidx,
    const unsigned short* __restrict__ W1fT, const float* __restrict__ b1f,
    const unsigned short* __restrict__ W2fT, const float* __restrict__ b2f,
    const unsigned short* __restrict__ W1wT, const float* __restrict__ b1w,
    const unsigned short* __restrict__ W2wT, const float* __restrict__ b2w,
    float* __restrict__ fout) {
  __shared__ __align__(16) unsigned short X1[32 * 96];    // [dxyz | feats | pad]
  __shared__ __align__(16) unsigned short Hh[32 * 128];   // hidden (reused by both branches)
  __shared__ __align__(16) unsigned short FP[32 * 128];   // f_prime (bf16)
  __shared__ __align__(16) unsigned short X2[32 * 160];   // [dxyz | centered f_prime | pad]
  __shared__ float MEAN[H_];

  const int lane = threadIdx.x;
  const int grp  = blockIdx.x;           // b*NPOINT + p
  const int b    = grp >> 10;

  const unsigned g = gidx[(size_t)grp * NSAMPLE_ + lane];
  const float cx = newxyz[(size_t)grp*3+0];
  const float cy = newxyz[(size_t)grp*3+1];
  const float cz = newxyz[(size_t)grp*3+2];
  const float* pxyz = xyz + ((size_t)b * N_ + g) * 3;
  const float dx = pxyz[0]-cx, dy = pxyz[1]-cy, dz = pxyz[2]-cz;

  {
    unsigned short* row = X1 + lane * 96;
    row[0] = f2bf(dx); row[1] = f2bf(dy); row[2] = f2bf(dz);
    const float* pf = feats + ((size_t)b * N_ + g) * C_;
#pragma unroll
    for (int c = 0; c < C_; ++c) row[3 + c] = f2bf(pf[c]);
#pragma unroll
    for (int c = 3 + C_; c < 96; ++c) row[c] = 0;
  }
  __syncthreads();

  gemm_relu(X1, W1fT, b1f, Hh, 96, 3, lane);     // h1 = relu(f_in @ W1f + b1f)
  __syncthreads();
  gemm_relu(Hh, W2fT, b2f, FP, 128, 4, lane);    // f' = relu(h1 @ W2f + b2f)
  __syncthreads();

  for (int c = lane; c < H_; c += 32) {          // mean over 32 samples per channel
    float s = 0.f;
#pragma unroll
    for (int m = 0; m < 32; ++m) s += bf2f(FP[m * H_ + c]);
    MEAN[c] = s * (1.f / 32.f);
  }
  __syncthreads();

  {
    unsigned short* r2 = X2 + lane * 160;
    r2[0] = f2bf(dx); r2[1] = f2bf(dy); r2[2] = f2bf(dz);
#pragma unroll
    for (int c = 0; c < H_; ++c) r2[3 + c] = f2bf(bf2f(FP[lane * H_ + c]) - MEAN[c]);
#pragma unroll
    for (int c = 3 + H_; c < 160; ++c) r2[c] = 0;
  }
  __syncthreads();

  gemm_relu(X2, W1wT, b1w, Hh, 160, 5, lane);    // h2 = relu(w_in @ W1w + b1w)
  __syncthreads();

  // alpha = sigmoid(h2 @ W2w + b2w); f_out = sum_s alpha * f'   (2x2-tile, 4 chains)
  const int nlo = lane & 15;
  const int mb  = (lane >> 4) << 3;
  float* outp = fout + (size_t)grp * H_;
  for (int nt = 0; nt < 8; nt += 2) {
    const int n0 = nt * 16 + nlo;
    const int n1 = n0 + 16;
    const float bv0 = b2w[n0];
    const float bv1 = b2w[n1];
    FragAcc a00, a10, a01, a11;
#pragma unroll
    for (int r = 0; r < 8; ++r) { a00.f[r]=0.f; a10.f[r]=0.f; a01.f[r]=0.f; a11.f[r]=0.f; }
#pragma unroll
    for (int kt = 0; kt < 4; ++kt) {
      v16bf x0 = load_a(Hh, 128, 0,  kt * 32, lane);
      v16bf x1 = load_a(Hh, 128, 16, kt * 32, lane);
      v16bf b0 = load_b(W2wT, 128, nt * 16,      kt * 32, lane);
      v16bf b1 = load_b(W2wT, 128, nt * 16 + 16, kt * 32, lane);
      a00.v = wmma_bf16(x0, b0, a00.v);
      a10.v = wmma_bf16(x1, b0, a10.v);
      a01.v = wmma_bf16(x0, b1, a01.v);
      a11.v = wmma_bf16(x1, b1, a11.v);
    }
    float p0 = 0.f, p1 = 0.f;
#pragma unroll
    for (int r = 0; r < 8; ++r) {
      float al00 = 1.f / (1.f + __expf(-(a00.f[r] + bv0)));
      float al10 = 1.f / (1.f + __expf(-(a10.f[r] + bv0)));
      float al01 = 1.f / (1.f + __expf(-(a01.f[r] + bv1)));
      float al11 = 1.f / (1.f + __expf(-(a11.f[r] + bv1)));
      p0 += al00 * bf2f(FP[(mb + r) * H_ + n0]);
      p0 += al10 * bf2f(FP[(16 + mb + r) * H_ + n0]);
      p1 += al01 * bf2f(FP[(mb + r) * H_ + n1]);
      p1 += al11 * bf2f(FP[(16 + mb + r) * H_ + n1]);
    }
    p0 += __shfl_xor(p0, 16, 32);                 // combine M and M+8 lane halves
    p1 += __shfl_xor(p1, 16, 32);
    if (lane < 16) { outp[n0] = p0; outp[n1] = p1; }
  }
}

// ------------------------------------------- launch ---------------------------------------
extern "C" void kernel_launch(void* const* d_in, const int* in_sizes, int n_in,
                              void* d_out, int out_size, void* d_ws, size_t ws_size,
                              hipStream_t stream) {
  const float* xyz   = (const float*)d_in[0];
  const float* feats = (const float*)d_in[1];
  const float* W1f   = (const float*)d_in[2];
  const float* b1f   = (const float*)d_in[3];
  const float* W2f   = (const float*)d_in[4];
  const float* b2f   = (const float*)d_in[5];
  const float* W1w   = (const float*)d_in[6];
  const float* b1w   = (const float*)d_in[7];
  const float* W2w   = (const float*)d_in[8];
  const float* b2w   = (const float*)d_in[9];

  float* out    = (float*)d_out;
  float* newxyz = out;                                   // (8,1024,3)
  float* fout   = out + (size_t)B_ * NPOINT_ * 3;        // (8,1024,128)

  char* ws = (char*)d_ws;
  size_t off = 0;
  unsigned* gidx = (unsigned*)(ws + off);  off += (size_t)B_ * NPOINT_ * NSAMPLE_ * sizeof(unsigned);
  unsigned short* W1fT = (unsigned short*)(ws + off);  off += (size_t)H_ * 96  * 2;
  unsigned short* W2fT = (unsigned short*)(ws + off);  off += (size_t)H_ * 128 * 2;
  unsigned short* W1wT = (unsigned short*)(ws + off);  off += (size_t)H_ * 160 * 2;
  unsigned short* W2wT = (unsigned short*)(ws + off);

  wprep_kernel<<<(H_*96  + 255)/256, 256, 0, stream>>>(W1f, W1fT, 67,  96);
  wprep_kernel<<<(H_*128 + 255)/256, 256, 0, stream>>>(W2f, W2fT, 128, 128);
  wprep_kernel<<<(H_*160 + 255)/256, 256, 0, stream>>>(W1w, W1wT, 131, 160);
  wprep_kernel<<<(H_*128 + 255)/256, 256, 0, stream>>>(W2w, W2wT, 128, 128);

  fps_kernel  <<<B_,            256, 0, stream>>>(xyz, newxyz);
  ballq_kernel<<<B_ * NPOINT_,  256, 0, stream>>>(xyz, newxyz, gidx);
  sa_compute_kernel<<<B_ * NPOINT_, 32, 0, stream>>>(xyz, feats, newxyz, gidx,
      W1fT, b1f, W2fT, b2f, W1wT, b1w, W2wT, b2w, fout);
}
